// BiLSTM_CRF_14405320311361
// MI455X (gfx1250) — matrix operand order repaired and checked by reading, not compiled, not run on hardware
//
#include <hip/hip_runtime.h>
#include <hip/hip_bf16.h>

typedef _Float16 v16h __attribute__((ext_vector_type(16)));
typedef _Float16 v8h  __attribute__((ext_vector_type(8)));
typedef _Float16 h2   __attribute__((ext_vector_type(2)));
typedef float    v8f  __attribute__((ext_vector_type(8)));

#define L_SENT 2048
#define E_DIM  256
#define HID    512
#define HB     256      // per-direction hidden
#define G4     1024     // 4*HB gate rows
#define TAGS   12
#define TPAD   16
#define START_TAG 10
#define STOP_TAG  11
#define NEGV  -10000.0f

__device__ __forceinline__ float sigf(float x) { return 1.0f / (1.0f + __expf(-x)); }

// ------------------------------------------------------------------
// Embedding gather + f32 -> f16 convert.  One thread per element.
// ------------------------------------------------------------------
__global__ void k_gather(const int* __restrict__ sent, const float* __restrict__ embed,
                         _Float16* __restrict__ xh) {
  int i = blockIdx.x * blockDim.x + threadIdx.x;   // over L*E
  int l = i >> 8;
  int e = i & 255;
  xh[i] = (_Float16)embed[(size_t)sent[l] * E_DIM + e];
}

// ------------------------------------------------------------------
// Pack W [Nrows][K] (row-major f32, gate-major) into the CDNA5 WMMA
// B-matrix layout (B = W^T), f16, N padded to multiple of 16 with 0.
// Layout: half index ((nn*ktiles + kk)*32 + lane)*16 + i  where
//   n = nn*16 + (lane&15);  k = kk*32 + (lane<16 ? 0 : 16) + i
// (contiguous k per lane half, same order as the A loads: VGPR v packs rows kbase+2v, kbase+2v+1)
// ------------------------------------------------------------------
__global__ void k_packB(const float* __restrict__ W, _Float16* __restrict__ out,
                        int Nrows, int K) {
  int tid  = blockIdx.x * blockDim.x + threadIdx.x;
  int i    = tid & 15;
  int lane = (tid >> 4) & 31;
  int kt   = tid >> 9;                 // nn*ktiles + kk
  int ktiles = K >> 5;
  int kk = kt % ktiles;
  int nn = kt / ktiles;
  int n  = nn * 16 + (lane & 15);
  int k  = kk * 32 + ((lane & 16) ? 16 : 0) + i;
  out[tid] = (n < Nrows) ? (_Float16)W[(size_t)n * K + k] : (_Float16)0.0f;
}

// ------------------------------------------------------------------
// Pack W_hh [1024][256] f32 into [k(128)][t(1024)] f16 pairs so the
// recurrence inner loop does coalesced b32 loads of (w[t][2k],w[t][2k+1]).
// ------------------------------------------------------------------
__global__ void k_packHH(const float* __restrict__ W, _Float16* __restrict__ out) {
  int tid = blockIdx.x * blockDim.x + threadIdx.x;   // 262144 per direction
  int hlf = tid & 1;
  int t   = (tid >> 1) & 1023;
  int k   = tid >> 11;
  out[tid] = (_Float16)W[(size_t)t * HB + 2 * k + hlf];
}

// ------------------------------------------------------------------
// Bias sums (b_ih + b_hh per direction) and padded output bias.
// ------------------------------------------------------------------
__global__ void k_bias(const float* __restrict__ bihf, const float* __restrict__ bhhf,
                       const float* __restrict__ bihb, const float* __restrict__ bhhb,
                       const float* __restrict__ bout,
                       float* __restrict__ biasf, float* __restrict__ biasb,
                       float* __restrict__ bo) {
  int t = threadIdx.x;
  if (t < G4) {
    biasf[t] = bihf[t] + bhhf[t];
    biasb[t] = bihb[t] + bhhb[t];
  }
  if (t < TPAD) bo[t] = (t < TAGS) ? bout[t] : 0.0f;
}

// ------------------------------------------------------------------
// WMMA GEMM:  C[M][N] = A[M][K](f16,row-major) * B(packed) + bias[N]
// One wave per 16x16 C tile; K-loop of v_wmma_f32_16x16x32_f16.
// A 16-bit layout: lane<16 holds K {kb..kb+15}
// with kb = kk*32; lane>=16 same rows, kb = kk*32+16.
// ------------------------------------------------------------------
__global__ __launch_bounds__(256)
void k_wmma_gemm(const _Float16* __restrict__ A,
                 const _Float16* __restrict__ Bp,
                 const float* __restrict__ bias,
                 float* __restrict__ C,
                 int M, int N, int K) {
  const int lane = threadIdx.x & 31;
  const int wave = threadIdx.x >> 5;
  const int ntiles = N >> 4;
  const int ktiles = K >> 5;
  const int tile = blockIdx.x * 8 + wave;
  const int mt = tile / ntiles;
  const int nt = tile - mt * ntiles;
  if (mt * 16 >= M) return;                       // uniform per wave: EXEC stays all-1
  const int lhi = lane >> 4;                      // 0 / 1
  const int l15 = lane & 15;
  const int row = mt * 16 + l15;

  float bv = bias[nt * 16 + l15];
  v8f c = { bv, bv, bv, bv, bv, bv, bv, bv };

  const _Float16* arow = A + (size_t)row * K + (lhi << 4);
  const _Float16* bp   = Bp + (size_t)nt * ktiles * 512 + (size_t)lane * 16;

  for (int kk = 0; kk < ktiles; ++kk) {
    v16h a = *(const v16h*)(arow + kk * 32);
    v16h b = *(const v16h*)(bp + (size_t)kk * 512);
    c = __builtin_amdgcn_wmma_f32_16x16x32_f16(false, a, false, b, (short)0, c,
                                               false, false);
  }

  float* crow = C + (size_t)(mt * 16 + lhi * 8) * N + nt * 16 + l15;
#pragma unroll
  for (int r = 0; r < 8; ++r) crow[(size_t)r * N] = c[r];
}

// ------------------------------------------------------------------
// Sequential BiLSTM recurrence.  One workgroup (32 waves) per direction,
// thread t owns gate row t.  h kept in LDS as f16 pairs (broadcast reads);
// W_hh streamed from L2 with coalesced b32 loads; packed f16 FMA dot.
// ------------------------------------------------------------------
__global__ __launch_bounds__(1024)
void k_lstm(const float* __restrict__ xg,        // [2][L][1024]
            const _Float16* __restrict__ whh,    // [2][128][1024] f16 pairs
            _Float16* __restrict__ hall) {       // [L][512]
  const int dir = blockIdx.x;
  const int t   = threadIdx.x;
  const float* xgd = xg + (size_t)dir * L_SENT * G4;
  const h2* wpk = (const h2*)(whh + (size_t)dir * 128 * G4 * 2);

  __shared__ float g_sh[G4];
  __shared__ __align__(4) _Float16 h_half[HB];
  if (t < 128) {
    h2 z = {};
    ((h2*)h_half)[t] = z;
  }
  float c = 0.0f;

  for (int s = 0; s < L_SENT; ++s) {
    const int step = dir ? (L_SENT - 1 - s) : s;
    __syncthreads();                               // h for this step is ready
    h2 acc = {};
    const h2* hp = (const h2*)h_half;
#pragma unroll 16
    for (int k = 0; k < 128; ++k) {
      h2 w  = wpk[(size_t)k * G4 + t];             // coalesced 128B/wave
      h2 hv = hp[k];                               // LDS broadcast
      acc = w * hv + acc;                          // v_pk_fma_f16
    }
    float g = xgd[(size_t)step * G4 + t] + (float)acc[0] + (float)acc[1];
    g_sh[t] = g;
    __syncthreads();
    if (t < HB) {
      float ig = sigf(g_sh[t]);
      float fg = sigf(g_sh[HB + t]);
      float gg = tanhf(g_sh[2 * HB + t]);
      float og = sigf(g_sh[3 * HB + t]);
      c = fg * c + ig * gg;
      float h = og * tanhf(c);
      h_half[t] = (_Float16)h;
      hall[(size_t)step * HID + dir * HB + t] = (_Float16)h;
    }
  }
}

// ------------------------------------------------------------------
// Viterbi DP + backtrack in a single wave32.  Lane n < 12 owns tag n.
// ------------------------------------------------------------------
__global__ void k_viterbi(const float* __restrict__ feats,   // [L][16]
                          const float* __restrict__ trans,   // [12][12]
                          int* __restrict__ bps,             // [L][12]
                          float* __restrict__ out) {         // [1 + L]
  const int lane = threadIdx.x;
  __shared__ float v[TAGS];
  float trow[TAGS];
  if (lane < TAGS) {
    v[lane] = (lane == START_TAG) ? 0.0f : NEGV;
    for (int p = 0; p < TAGS; ++p) trow[p] = trans[lane * TAGS + p];
  }
  __syncthreads();
  for (int t = 0; t < L_SENT; ++t) {
    float best = -3.0e38f;
    int bi = 0;
    if (lane < TAGS) {
      for (int p = 0; p < TAGS; ++p) {
        float s = trow[p] + v[p];
        if (s > best) { best = s; bi = p; }        // strict > : first argmax wins
      }
    }
    __syncthreads();
    if (lane < TAGS) {
      v[lane] = best + feats[t * TPAD + lane];
      bps[t * TAGS + lane] = bi;
    }
    __syncthreads();
  }
  if (lane == 0) {
    float best = -3.0e38f;
    int bt = 0;
    for (int n = 0; n < TAGS; ++n) {
      float s = v[n] + trans[STOP_TAG * TAGS + n];
      if (s > best) { best = s; bt = n; }
    }
    out[0] = best;
    int tag = bt;
    for (int t = L_SENT - 1; t >= 0; --t) {
      out[1 + t] = (float)tag;
      tag = bps[t * TAGS + tag];
    }
  }
}

// ------------------------------------------------------------------
extern "C" void kernel_launch(void* const* d_in, const int* in_sizes, int n_in,
                              void* d_out, int out_size, void* d_ws, size_t ws_size,
                              hipStream_t stream) {
  (void)in_sizes; (void)n_in; (void)out_size; (void)ws_size;
  const int*   sent   = (const int*)d_in[0];
  const float* embed  = (const float*)d_in[1];
  const float* w_ih_f = (const float*)d_in[2];
  const float* w_hh_f = (const float*)d_in[3];
  const float* b_ih_f = (const float*)d_in[4];
  const float* b_hh_f = (const float*)d_in[5];
  const float* w_ih_b = (const float*)d_in[6];
  const float* w_hh_b = (const float*)d_in[7];
  const float* b_ih_b = (const float*)d_in[8];
  const float* b_hh_b = (const float*)d_in[9];
  const float* w_out  = (const float*)d_in[10];
  const float* b_out  = (const float*)d_in[11];
  const float* trans  = (const float*)d_in[12];
  float* out = (float*)d_out;

  char* ws = (char*)d_ws;
  size_t off = 0;
  auto alloc = [&](size_t bytes) -> void* {
    void* p = ws + off;
    off = (off + bytes + 255) & ~(size_t)255;
    return p;
  };

  _Float16* xh    = (_Float16*)alloc((size_t)L_SENT * E_DIM * 2);       // 1 MB
  _Float16* pIHf  = (_Float16*)alloc((size_t)E_DIM * G4 * 2);           // 512 KB
  _Float16* pIHb  = (_Float16*)alloc((size_t)E_DIM * G4 * 2);           // 512 KB
  _Float16* pOut  = (_Float16*)alloc((size_t)TPAD * HID * 2);           // 16 KB
  _Float16* pHH   = (_Float16*)alloc((size_t)2 * 128 * G4 * 2 * 2);     // 1 MB
  float*    biasf = (float*)alloc((size_t)G4 * 4);
  float*    biasb = (float*)alloc((size_t)G4 * 4);
  float*    bo    = (float*)alloc((size_t)TPAD * 4);
  float*    xgbuf = (float*)alloc((size_t)2 * L_SENT * G4 * 4);         // 16 MB
  _Float16* hall  = (_Float16*)alloc((size_t)L_SENT * HID * 2);         // 2 MB
  float*    feats = (float*)alloc((size_t)L_SENT * TPAD * 4);           // 128 KB
  int*      bps   = (int*)alloc((size_t)L_SENT * TAGS * 4);             // 96 KB

  // 1) gather + convert
  k_gather<<<(L_SENT * E_DIM) / 256, 256, 0, stream>>>(sent, embed, xh);

  // 2) weight packing (independent of gather)
  k_packB<<<(E_DIM * G4) / 256, 256, 0, stream>>>(w_ih_f, pIHf, G4, E_DIM);
  k_packB<<<(E_DIM * G4) / 256, 256, 0, stream>>>(w_ih_b, pIHb, G4, E_DIM);
  k_packB<<<(TPAD * HID) / 256, 256, 0, stream>>>(w_out, pOut, TAGS, HID);
  k_packHH<<<(128 * G4 * 2) / 256, 256, 0, stream>>>(w_hh_f, pHH);
  k_packHH<<<(128 * G4 * 2) / 256, 256, 0, stream>>>(w_hh_b, pHH + (size_t)128 * G4 * 2);
  k_bias<<<1, 1024, 0, stream>>>(b_ih_f, b_hh_f, b_ih_b, b_hh_b, b_out,
                                 biasf, biasb, bo);

  // 3) input-projection GEMMs (WMMA):  xg = x @ W_ih^T + (b_ih + b_hh)
  const int xg_blocks = (L_SENT / 16) * (G4 / 16) / 8;   // 1024
  k_wmma_gemm<<<xg_blocks, 256, 0, stream>>>(xh, pIHf, biasf, xgbuf,
                                             L_SENT, G4, E_DIM);
  k_wmma_gemm<<<xg_blocks, 256, 0, stream>>>(xh, pIHb, biasb,
                                             xgbuf + (size_t)L_SENT * G4,
                                             L_SENT, G4, E_DIM);

  // 4) sequential recurrence, forward + backward concurrently
  k_lstm<<<2, 1024, 0, stream>>>(xgbuf, pHH, hall);

  // 5) output projection (WMMA, N padded 12 -> 16)
  k_wmma_gemm<<<(L_SENT / 16) / 8, 256, 0, stream>>>(hall, pOut, bo, feats,
                                                     L_SENT, TPAD, HID);

  // 6) Viterbi decode
  k_viterbi<<<1, 32, 0, stream>>>(feats, trans, bps, out);
}
